// ImplicitLayer_17119739642034
// MI455X (gfx1250) — matrix operand order repaired and checked by reading, not compiled, not run on hardware
//
#include <hip/hip_runtime.h>
#include <hip/hip_bf16.h>

typedef __attribute__((ext_vector_type(16))) _Float16 v16h;
typedef __attribute__((ext_vector_type(8)))  float    v8f;

#define WAVES_PER_BLOCK 8
#define ROWS_PER_WAVE   16
#define ZPAD_STRIDE     68   // 64 + 4 pad: float4-aligned rows for the final store

__device__ __forceinline__ float fast_tanh(float v) {
#if __has_builtin(__builtin_amdgcn_tanhf)
    return __builtin_amdgcn_tanhf(v);          // v_tanh_f32: 1 TRANS op
#else
    float e = __expf(2.0f * v);
    return 1.0f - 2.0f * __builtin_amdgcn_rcpf(e + 1.0f);
#endif
}

// full wave32 sum, pure VALU (DPP rotate-add within rows of 16, permlanex16 across)
__device__ __forceinline__ float wave_sum32(float v) {
#if __has_builtin(__builtin_amdgcn_update_dpp) && __has_builtin(__builtin_amdgcn_permlanex16)
    int t;
    t = __builtin_amdgcn_update_dpp(0, __float_as_int(v), 0x128, 0xf, 0xf, true); // row_ror:8
    v += __int_as_float(t);
    t = __builtin_amdgcn_update_dpp(0, __float_as_int(v), 0x124, 0xf, 0xf, true); // row_ror:4
    v += __int_as_float(t);
    t = __builtin_amdgcn_update_dpp(0, __float_as_int(v), 0x122, 0xf, 0xf, true); // row_ror:2
    v += __int_as_float(t);
    t = __builtin_amdgcn_update_dpp(0, __float_as_int(v), 0x121, 0xf, 0xf, true); // row_ror:1
    v += __int_as_float(t);
    t = __builtin_amdgcn_permlanex16(__float_as_int(v), __float_as_int(v),
                                     0x76543210, 0xfedcba98, false, false);       // half swap
    v += __int_as_float(t);
    return v;
#else
    #pragma unroll
    for (int off = 16; off > 0; off >>= 1) v += __shfl_xor(v, off, 32);
    return v;
#endif
}

__global__ __launch_bounds__(256, 2)
void deq_wmma_kernel(const float* __restrict__ u,
                     const float* __restrict__ Aw,   // [64,64]
                     const float* __restrict__ Ab,   // [64]
                     const float* __restrict__ Bw,   // [64,32]
                     const float* __restrict__ Bb,   // [64]
                     float* __restrict__ out)        // [batch,64]
{
    __shared__ float sAw[64 * 64];
    __shared__ float sBw[64 * 32];
    __shared__ float sAb[64];
    __shared__ float sBb[64];
    __shared__ float sZ[WAVES_PER_BLOCK][ROWS_PER_WAVE][ZPAD_STRIDE];

    const int tid = threadIdx.x;

    // ---- one-time cooperative, vectorized weight broadcast into LDS ----
    {
        const float4* Aw4 = (const float4*)Aw;
        const float4* Bw4 = (const float4*)Bw;
        float4* sAw4 = (float4*)sAw;
        float4* sBw4 = (float4*)sBw;
        #pragma unroll
        for (int i = 0; i < 4; ++i) sAw4[tid + 256 * i] = Aw4[tid + 256 * i];
        #pragma unroll
        for (int i = 0; i < 2; ++i) sBw4[tid + 256 * i] = Bw4[tid + 256 * i];
        if (tid < 64) { sAb[tid] = Ab[tid]; sBb[tid] = Bb[tid]; }
    }
    __syncthreads();

    const int wave = tid >> 5;
    const int lane = tid & 31;
    const int half = lane >> 4;
    const int l16  = lane & 15;
    const long row0 = (long)(blockIdx.x * WAVES_PER_BLOCK + wave) * ROWS_PER_WAVE;

    float (*zrow)[ZPAD_STRIDE] = sZ[wave];   // wave-private slab (final store only)

    // ---- A_w as WMMA A-operand with K columns permuted by
    //      phi(32kc+16h+s) = 16kc + 32(s>>3) + 8h + (s&7)
    //      => each operand = two runs of 8 f32 from row jrow, bases 16kc+8h and 16kc+8h+32 ----
    v16h awA[4][2];
    v16h bwA[4];      // B_w (natural K): two contiguous runs of 8
    #pragma unroll
    for (int tj = 0; tj < 4; ++tj) {
        const int jrow = tj * 16 + l16;                 // output feature row
        #pragma unroll
        for (int kc = 0; kc < 2; ++kc) {
            const float* src = &sAw[jrow * 64 + kc * 16 + 8 * half];
            #pragma unroll
            for (int s = 0; s < 16; ++s)
                awA[tj][kc][s] = (_Float16)src[(s & 7) + 32 * (s >> 3)];
        }
        {
            const float* s0 = &sBw[jrow * 32 + 16 * half];       // slots 0..7
            const float* s1 = &sBw[jrow * 32 + 16 * half + 8];   // slots 8..15
            #pragma unroll
            for (int s = 0; s < 8; ++s) {
                bwA[tj][s]     = (_Float16)s0[s];
                bwA[tj][8 + s] = (_Float16)s1[s];
            }
        }
    }

    // ---- u^T as WMMA B-operand: 16 contiguous floats per lane, direct loads ----
    v16h uB;
    {
        const float* up = u + (row0 + l16) * 32 + 16 * half;
        #pragma unroll
        for (int i = 0; i < 4; ++i) {
            const float4 q = ((const float4*)up)[i];
            uB[4 * i + 0] = (_Float16)q.x;
            uB[4 * i + 1] = (_Float16)q.y;
            uB[4 * i + 2] = (_Float16)q.z;
            uB[4 * i + 3] = (_Float16)q.w;
        }
    }

    // ---- constant term c' = (A_b + B_b)[j] + (B_w @ u^T)[j,m], transposed D layout ----
    v8f cAcc[4];
    #pragma unroll
    for (int tj = 0; tj < 4; ++tj) {
        const float4 a0 = *(const float4*)&sAb[tj * 16 + 8 * half];
        const float4 a1 = *(const float4*)&sAb[tj * 16 + 8 * half + 4];
        const float4 b0 = *(const float4*)&sBb[tj * 16 + 8 * half];
        const float4 b1 = *(const float4*)&sBb[tj * 16 + 8 * half + 4];
        cAcc[tj][0] = a0.x + b0.x; cAcc[tj][1] = a0.y + b0.y;
        cAcc[tj][2] = a0.z + b0.z; cAcc[tj][3] = a0.w + b0.w;
        cAcc[tj][4] = a1.x + b1.x; cAcc[tj][5] = a1.y + b1.y;
        cAcc[tj][6] = a1.z + b1.z; cAcc[tj][7] = a1.w + b1.w;
        cAcc[tj] = __builtin_amdgcn_wmma_f32_16x16x32_f16(
            false, bwA[tj], false, uB, (short)0, cAcc[tj], false, false);
    }

    // ---- f applied in place; residual sums folded in.
    //      zB[kc] slot s  <-  fx[2*(s>>3)+kc][s&7]   (pure in-lane remap!) ----
    auto apply_f = [&](v8f fx[4], float& d2, float& f2) {
        v16h zB[2];
        #pragma unroll
        for (int kc = 0; kc < 2; ++kc)
            #pragma unroll
            for (int s = 0; s < 16; ++s)
                zB[kc][s] = (_Float16)fx[2 * (s >> 3) + kc][s & 7];
        d2 = 0.0f; f2 = 0.0f;
        #pragma unroll
        for (int tj = 0; tj < 4; ++tj) {
            v8f acc = cAcc[tj];
            acc = __builtin_amdgcn_wmma_f32_16x16x32_f16(
                false, awA[tj][0], false, zB[0], (short)0, acc, false, false);
            acc = __builtin_amdgcn_wmma_f32_16x16x32_f16(
                false, awA[tj][1], false, zB[1], (short)0, acc, false, false);
            #pragma unroll
            for (int r = 0; r < 8; ++r) {
                const float old = fx[tj][r];
                const float nv  = fast_tanh(acc[r]);
                const float d   = nv - old;
                d2 += d * d;
                f2 += nv * nv;
                fx[tj][r] = nv;
            }
        }
    };

    // ---- fixed-point iteration: x0 = 0, fx = f(0) = tanh(c') ----
    v8f fx[4];
    float d2 = 0.0f, f2 = 0.0f;
    #pragma unroll
    for (int tj = 0; tj < 4; ++tj)
        #pragma unroll
        for (int r = 0; r < 8; ++r) {
            const float nv = fast_tanh(cAcc[tj][r]);
            fx[tj][r] = nv;
            d2 += nv * nv;           // x0 = 0 -> diff == fx
            f2 += nv * nv;
        }

    const float eps = 1e-3f;
    int k = 0;
    for (;;) {
        const float D2 = wave_sum32(d2);
        const float F2 = wave_sum32(f2);
        int cont = (k < 200) &&
                   (__builtin_sqrtf(D2) >= eps * (1e-5f + __builtin_sqrtf(F2)));
        if (!__builtin_amdgcn_readfirstlane(cont))   // scalar, wave-uniform exit
            break;
        ++k;
        apply_f(fx, d2, f2);
    }

    // module applies f once more
    apply_f(fx, d2, f2);

    // ---- one-time store: de-transpose into LDS with wide DS stores, then
    //      fully-coalesced b128 global stores over the contiguous 16x64 tile ----
    #pragma unroll
    for (int tj = 0; tj < 4; ++tj) {
        float4 lo, hi;
        lo.x = fx[tj][0]; lo.y = fx[tj][1]; lo.z = fx[tj][2]; lo.w = fx[tj][3];
        hi.x = fx[tj][4]; hi.y = fx[tj][5]; hi.z = fx[tj][6]; hi.w = fx[tj][7];
        *(float4*)&zrow[l16][tj * 16 + 8 * half]     = lo;   // z[l16][tj*16+8h+0..3]
        *(float4*)&zrow[l16][tj * 16 + 8 * half + 4] = hi;   // z[l16][tj*16+8h+4..7]
    }
    {
        float* obase = out + row0 * 64;                // contiguous 16x64 f32 tile
        #pragma unroll
        for (int i = 0; i < 8; ++i) {
            const int f = i * 128 + lane * 4;
            const int rr = f >> 6, cc = f & 63;
            *(float4*)(obase + f) = *(const float4*)&zrow[rr][cc];
        }
    }
}

extern "C" void kernel_launch(void* const* d_in, const int* in_sizes, int n_in,
                              void* d_out, int out_size, void* d_ws, size_t ws_size,
                              hipStream_t stream) {
    const float* u  = (const float*)d_in[0];   // [batch, 32]
    const float* Aw = (const float*)d_in[1];   // [64, 64]
    const float* Ab = (const float*)d_in[2];   // [64]
    const float* Bw = (const float*)d_in[3];   // [64, 32]
    const float* Bb = (const float*)d_in[4];   // [64]
    float* out = (float*)d_out;

    const int batch = in_sizes[0] / 32;
    const int rows_per_block = WAVES_PER_BLOCK * ROWS_PER_WAVE;   // 128
    const int blocks = batch / rows_per_block;                    // 8192 for 1M rows

    deq_wmma_kernel<<<dim3(blocks), dim3(256), 0, stream>>>(u, Aw, Ab, Bw, Bb, out);
}
